// TMPNN_46127948759692
// MI455X (gfx1250) — matrix operand order repaired and checked by reading, not compiled, not run on hardware
//
#include <hip/hip_runtime.h>

// ---------------------------------------------------------------------------
// TaylorMap degree-2, 7-step recurrence, fully fused on-chip (MI455X gfx1250).
//   x <- [1, x, x (x) x] @ W   (state N=64, NPOLY=4161), 7 steps; out = x[:,48:64]
// Roofline: ~1.2e11 FLOP vs ~10 MB minimal traffic -> ~60x compute-bound at
// f16-WMMA peak; so never materialize features, synthesize A-tiles in regs:
// for k-tile t (32 quad features), first factor index i=t>>1 is constant ->
//   A_frag(t) = state_frag(t&1) * splat(x[m, t>>1])   (8x v_pk_mul_f16 / WMMA)
// This round: 2 row-tiles per wave per B fragment (2x arithmetic intensity on
// the L2 B stream, independent WMMA chains), chunked v4h scalar preloads
// (kills per-tile s_wait_dscnt), lazy affine-fragment loads (lower pressure).
// ---------------------------------------------------------------------------

typedef _Float16 v16h __attribute__((ext_vector_type(16)));
typedef _Float16 v8h  __attribute__((ext_vector_type(8)));
typedef _Float16 v4h  __attribute__((ext_vector_type(4)));
typedef float    v8f  __attribute__((ext_vector_type(8)));

#define STEPS     7
#define NIN       48
#define NST       64
#define KT_Q      128            // quadratic k-tiles: K = 4096
#define KT_A      4              // affine k-tiles:    K = 128 = [1, x(64), 0(63)]
#define KT        (KT_Q + KT_A)  // 132
#define NSLAB     4              // 4 slabs of 16 output columns
#define ROWS_WG   128            // batch rows per workgroup (8 row-tiles)
#define XH_STRIDE 72             // halves; 144 B row stride (16B aligned, bank-spread)
#define AF_STRIDE 136            // halves; 272 B row stride (16B aligned, bank-spread)

static __device__ __forceinline__ v16h vsplat16(_Float16 s) {
    v16h v;
#pragma unroll
    for (int i = 0; i < 16; ++i) v[i] = s;
    return v;
}

#define WMMA_F16(A, B, C) \
    __builtin_amdgcn_wmma_f32_16x16x32_f16(false, (A), false, (B), (short)0, (C), false, false)

// ---------------------------------------------------------------------------
// Pre-pass: pack W (4161x64 f32) into f16 B-matrix fragments.
// Fragment f = t*4 + s (k-tile t, col-slab s): 32 lanes x 16 halves.
// B 32x16 f16 layout: lane = n + 16*((k>>3)&1); dword v holds K = 8h+2v, +1, plus 8 if v>=4 (h = lane>>4).
// ---------------------------------------------------------------------------
__global__ void wfrag_build(const float* __restrict__ W, _Float16* __restrict__ wf) {
    const int f    = blockIdx.x;       // 0..527
    const int lane = threadIdx.x;      // 0..31
    const int t    = f >> 2;
    const int s    = f & 3;
    const int n    = s * 16 + (lane & 15);
    _Float16* out  = wf + (size_t)f * 512 + (size_t)lane * 16;
#pragma unroll
    for (int v = 0; v < 8; ++v) {
#pragma unroll
        for (int hh = 0; hh < 2; ++hh) {
            const int kl = (lane >> 4) * 8 + 2 * v + hh + ((v >= 4) ? 8 : 0);   // 0..31 within tile
            float val;
            if (t < KT_Q) {
                const int q = t * 32 + kl;                  // quad feature i*64+j
                val = W[(size_t)(1 + NST + q) * NST + n];   // rows 65..4160
            } else {
                const int ka = (t - KT_Q) * 32 + kl;        // 0..127 affine feat
                val = (ka <= NST) ? W[(size_t)ka * NST + n] // ka==0 -> bias row 0
                                  : 0.0f;                   // zero padding
            }
            out[v * 2 + hh] = (_Float16)val;
        }
    }
}

// ---------------------------------------------------------------------------
// Main fused kernel: 512 threads = 16 waves; wave (r,s) owns TWO 16x16 tiles
// (row-tiles 2r, 2r+1) so each B fragment feeds two independent WMMA chains.
// ---------------------------------------------------------------------------
__global__ void __launch_bounds__(512)
taylor7_kernel(const float* __restrict__ X, const v16h* __restrict__ wf,
               float* __restrict__ out) {
    __shared__ _Float16 xh[ROWS_WG * XH_STRIDE];   // state, row-major f16
    __shared__ _Float16 af[ROWS_WG * AF_STRIDE];   // affine features [1, x, 0..]

    const int tid  = threadIdx.x;
    const int lane = tid & 31;
    const int wave = tid >> 5;      // 0..15
    const int r    = wave >> 2;     // row-tile pair 0..3 (rows r*32 .. r*32+31)
    const int s    = wave & 3;      // col slab 0..3
    const int mrow = lane & 15;     // A-matrix row owned by this lane
    const int g    = lane >> 4;     // lane group (selects K sub-ranges)
    const int rowbase = blockIdx.x * ROWS_WG;

    // ---- load X tile, pad latent cols with zeros, convert to f16 ----
    for (int idx = tid; idx < ROWS_WG * NST; idx += 512) {
        const int mm = idx >> 6, cc = idx & 63;
        const float v = (cc < NIN) ? X[(size_t)(rowbase + mm) * NIN + cc] : 0.0f;
        const _Float16 h16 = (_Float16)v;
        xh[mm * XH_STRIDE + cc]     = h16;
        af[mm * AF_STRIDE + 1 + cc] = h16;
    }
    for (int mm = tid; mm < ROWS_WG; mm += 512) {
        af[mm * AF_STRIDE] = (_Float16)1.0f;       // constant feature
#pragma unroll
        for (int z = 1 + NST; z < 128; ++z)        // zero padding (written once)
            af[mm * AF_STRIDE + z] = (_Float16)0.0f;
    }
    __syncthreads();

    const _Float16* xrowA = &xh[(r * 32 + mrow) * XH_STRIDE];
    const _Float16* xrowB = xrowA + 16 * XH_STRIDE;
    const _Float16* arowA = &af[(r * 32 + mrow) * AF_STRIDE];
    const _Float16* arowB = arowA + 16 * AF_STRIDE;
    const v16h*     wsl   = wf + (size_t)s * 32 + lane;   // + t*128 per k-tile

    for (int step = 0; step < STEPS; ++step) {
        // --- gather state A fragments (16x32 f16) for both row-tiles ---
        v16h sfA[2], sfB[2];
#pragma unroll
        for (int hb = 0; hb < 2; ++hb) {             // state cols [0..31],[32..63]
            {
                const v8h lo = *(const v8h*)&xrowA[hb * 32 + g * 8];
                const v8h hi = *(const v8h*)&xrowA[hb * 32 + 16 + g * 8];
                v16h fv;
#pragma unroll
                for (int e = 0; e < 8; ++e) { fv[e] = lo[e]; fv[8 + e] = hi[e]; }
                sfA[hb] = fv;
            }
            {
                const v8h lo = *(const v8h*)&xrowB[hb * 32 + g * 8];
                const v8h hi = *(const v8h*)&xrowB[hb * 32 + 16 + g * 8];
                v16h fv;
#pragma unroll
                for (int e = 0; e < 8; ++e) { fv[e] = lo[e]; fv[8 + e] = hi[e]; }
                sfB[hb] = fv;
            }
        }

        v8f accA, accB;
#pragma unroll
        for (int e = 0; e < 8; ++e) { accA[e] = 0.0f; accB[e] = 0.0f; }

        // --- quadratic part: 128 k-tiles in chunks of 8; one B feeds 2 WMMA ---
        v16h bcur = wsl[0];
#pragma unroll 1
        for (int tc = 0; tc < KT_Q / 8; ++tc) {
            // 4 first-factor scalars cover 8 tiles; one ds_load_b64 each
            const v4h xsA = *(const v4h*)&xrowA[tc * 4];
            const v4h xsB = *(const v4h*)&xrowB[tc * 4];
#pragma unroll
            for (int u = 0; u < 8; ++u) {
                const int t = tc * 8 + u;
                const v16h bnext = wsl[(size_t)(t + 1) * 128]; // t=127 -> affine[0]
                const v16h aA = sfA[t & 1] * vsplat16(xsA[u >> 1]);
                accA = WMMA_F16(aA, bcur, accA);
                const v16h aB = sfB[t & 1] * vsplat16(xsB[u >> 1]);
                accB = WMMA_F16(aB, bcur, accB);
                bcur = bnext;
            }
        }

        // --- affine tail: fragments loaded lazily (lower live pressure) ---
#pragma unroll
        for (int j = 0; j < KT_A; ++j) {
            const v16h bb = (j == 0) ? bcur : wsl[(size_t)(KT_Q + j) * 128];
            {
                const v8h lo = *(const v8h*)&arowA[j * 32 + g * 8];
                const v8h hi = *(const v8h*)&arowA[j * 32 + 16 + g * 8];
                v16h fv;
#pragma unroll
                for (int e = 0; e < 8; ++e) { fv[e] = lo[e]; fv[8 + e] = hi[e]; }
                accA = WMMA_F16(fv, bb, accA);
            }
            {
                const v8h lo = *(const v8h*)&arowB[j * 32 + g * 8];
                const v8h hi = *(const v8h*)&arowB[j * 32 + 16 + g * 8];
                v16h fv;
#pragma unroll
                for (int e = 0; e < 8; ++e) { fv[e] = lo[e]; fv[8 + e] = hi[e]; }
                accB = WMMA_F16(fv, bb, accB);
            }
        }

        if (step < STEPS - 1) {
            __syncthreads();                 // all reads of old state complete
            // C/D layout: dword c -> M = c (lanes 0-15) or 8+c (lanes 16-31),
            //             N = lane&15
            const int nn = s * 16 + (lane & 15);
#pragma unroll
            for (int c = 0; c < 8; ++c) {
                const int moA = r * 32 + (g ? 8 : 0) + c;
                const int moB = moA + 16;
                const _Float16 hA = (_Float16)accA[c];
                const _Float16 hB = (_Float16)accB[c];
                xh[moA * XH_STRIDE + nn]     = hA;
                af[moA * AF_STRIDE + 1 + nn] = hA;
                xh[moB * XH_STRIDE + nn]     = hB;
                af[moB * AF_STRIDE + 1 + nn] = hB;
            }
            __syncthreads();
        } else if (s == 3) {
            // final step: slab 3 == state cols 48..63 == targets; store f32
#pragma unroll
            for (int c = 0; c < 8; ++c) {
                const int moA = r * 32 + (g ? 8 : 0) + c;
                out[(size_t)(rowbase + moA) * 16 + (lane & 15)]      = accA[c];
                out[(size_t)(rowbase + moA + 16) * 16 + (lane & 15)] = accB[c];
            }
        }
    }
}

extern "C" void kernel_launch(void* const* d_in, const int* in_sizes, int n_in,
                              void* d_out, int out_size, void* d_ws, size_t ws_size,
                              hipStream_t stream) {
    const float* X = (const float*)d_in[0];   // (32768, 48) f32
    const float* W = (const float*)d_in[1];   // (4161, 64)  f32
    float* outp    = (float*)d_out;           // (32768, 16) f32
    _Float16* wfrag = (_Float16*)d_ws;        // needs 528*512*2 = 540,672 B

    const int batch = in_sizes[0] / NIN;      // 32768

    // pack W into f16 B-fragments (deterministic, recomputed every call)
    wfrag_build<<<KT * NSLAB, 32, 0, stream>>>(W, wfrag);
    // fused 7-step recurrence, 128 batch rows per workgroup
    taylor7_kernel<<<batch / ROWS_WG, 512, 0, stream>>>(X, (const v16h*)wfrag, outp);
}
